// STU_62955630624947
// MI455X (gfx1250) — matrix operand order repaired and checked by reading, not compiled, not run on hardware
//
#include <hip/hip_runtime.h>
#include <hip/hip_bf16.h>

// STU forward for MI455X (gfx1250, wave32, WMMA).
// Math: plus+minus == 2 * sum_{even lag tau<=t} phi_proj[tau,d]*x_proj[b,t-tau,d]
//  -> two half-rate causal convs (per parity) with g[m,d]=2*phi_proj[2m,d].
// Conv = per-channel lower-triangular Toeplitz matmul with
// v_wmma_f32_16x16x32_bf16; Toeplitz pairs are pre-packed in LDS so every
// A-fragment VGPR is a single aligned ds_load_b32, and one B fragment is
// shared across 4 accumulator tiles per wave.

#define B_ 4
#define L_ 8192
#define D_ 1024
#define K_ 24
#define MH_ 4096            // half-rate length L/2
#define NTILE_ 256          // MH_/16 output row tiles per channel
#define PAD_ 128            // front zero pad (covers lag >= -63 in 4-tile groups)

typedef __attribute__((ext_vector_type(16))) __bf16 v16bf;
typedef __attribute__((ext_vector_type(8)))  float  v8f;

union FragAB { unsigned int u[8]; v16bf v; };

__device__ __forceinline__ unsigned short f2bf(float f) {
  unsigned int u = __float_as_uint(f);
  u += 0x7FFFu + ((u >> 16) & 1u);       // round-to-nearest-even
  return (unsigned short)(u >> 16);
}

// ---------------------------------------------------------------------------
// Kernel A: g_even[d][m] = bf16( 2 * sum_k phi[2m,k] * M_filters[k,d] )
// ---------------------------------------------------------------------------
__global__ __launch_bounds__(256) void stu_filter_kernel(
    const float* __restrict__ phi,        // [L_, K_]
    const float* __restrict__ Mf,         // [K_, D_]
    unsigned short* __restrict__ g_even)  // [D_, MH_] bf16
{
  int idx = blockIdx.x * 256 + threadIdx.x;   // d*MH_ + m
  int d = idx >> 12;
  int m = idx & (MH_ - 1);
  const float* prow = phi + (size_t)(2 * m) * K_;
  float acc = 0.f;
#pragma unroll
  for (int k = 0; k < K_; ++k) acc += prow[k] * Mf[k * D_ + d];
  g_even[idx] = f2bf(2.0f * acc);
}

// ---------------------------------------------------------------------------
// Kernel B: x_proj = x @ M_inputs, epilogue writes bf16 into parity-permuted
// layout u_perm[((e*8) + (b*2+r)) * MH_ + m], where t = 2m+r.
// Workgroup: 256 thr (8 waves), tile M=128 x N=64, K-step 32.
// ---------------------------------------------------------------------------
__global__ __launch_bounds__(256) void stu_xproj_kernel(
    const float* __restrict__ x,          // [B_*L_, D_]
    const float* __restrict__ Mi,         // [D_, D_]
    unsigned short* __restrict__ u_perm)  // [D_][8][MH_] bf16
{
  __shared__ unsigned short shA[128][34];   // A tile (M x K) bf16, padded
  __shared__ unsigned short shBT[64][34];   // B tile transposed (N x K) bf16

  const int tid   = threadIdx.x;
  const int lane  = tid & 31;
  const int w     = tid >> 5;
  const int waveM = w >> 2;                 // 0..1  (64 rows each)
  const int waveN = w & 3;                  // 0..3  (16 cols each)
  const int mTile = blockIdx.y * 128;
  const int nTile = blockIdx.x * 64;

  v8f acc[4];
#pragma unroll
  for (int s = 0; s < 4; ++s)
#pragma unroll
    for (int v = 0; v < 8; ++v) acc[s][v] = 0.f;

  const int kbA = (lane >= 16) ? 8 : 0;     // unused; A frag also uses kbB
  const int kbB = (lane >= 16) ? 16 : 0;    // B-frag K base (contiguous k = 16h + s)

  for (int kk = 0; kk < D_; kk += 32) {
    // stage A: 128x32 fp32 -> bf16
#pragma unroll
    for (int j = 0; j < 4; ++j) {
      int idx = tid + j * 256;              // 0..1023 float4 chunks
      int row = idx >> 3, q = idx & 7;
      float4 f = *(const float4*)(x + (size_t)(mTile + row) * D_ + kk + q * 4);
      shA[row][q * 4 + 0] = f2bf(f.x);
      shA[row][q * 4 + 1] = f2bf(f.y);
      shA[row][q * 4 + 2] = f2bf(f.z);
      shA[row][q * 4 + 3] = f2bf(f.w);
    }
    // stage B transposed: 32x64 fp32 -> bf16 [n][k]
#pragma unroll
    for (int j = 0; j < 2; ++j) {
      int idx = tid + j * 256;              // 0..511 float4 chunks
      int krow = idx >> 4, q = idx & 15;
      float4 f = *(const float4*)(Mi + (size_t)(kk + krow) * D_ + nTile + q * 4);
      shBT[q * 4 + 0][krow] = f2bf(f.x);
      shBT[q * 4 + 1][krow] = f2bf(f.y);
      shBT[q * 4 + 2][krow] = f2bf(f.z);
      shBT[q * 4 + 3][krow] = f2bf(f.w);
    }
    __syncthreads();

    FragAB bfr;
    const int ncol = waveN * 16 + (lane & 15);
#pragma unroll
    for (int v = 0; v < 8; ++v)
      bfr.u[v] = *(const unsigned int*)&shBT[ncol][kbB + 2 * v];

#pragma unroll
    for (int s = 0; s < 4; ++s) {
      FragAB afr;
      const int row = waveM * 64 + s * 16 + (lane & 15);
#pragma unroll
      for (int v = 0; v < 8; ++v) {
        int K = kbB + 2 * v;
        afr.u[v] = *(const unsigned int*)&shA[row][K];
      }
      acc[s] = __builtin_amdgcn_wmma_f32_16x16x32_bf16(
          false, afr.v, false, bfr.v, (short)0, acc[s], false, false);
    }
    __syncthreads();
  }

  // epilogue: C/D layout -> permuted bf16 store
  const int e = nTile + waveN * 16 + (lane & 15);
#pragma unroll
  for (int s = 0; s < 4; ++s) {
    const int rbase = mTile + waveM * 64 + s * 16 + ((lane >> 4) << 3);
#pragma unroll
    for (int v = 0; v < 8; ++v) {
      int gm = rbase + v;                   // b*L_ + t
      int b = gm >> 13;
      int t = gm & (L_ - 1);
      int m = t >> 1, r = t & 1;
      u_perm[(((size_t)(e << 3) + (b << 1) + r) << 12) + m] = f2bf(acc[s][v]);
    }
  }
}

// ---------------------------------------------------------------------------
// Kernel C: per-channel lower-triangular Toeplitz conv via WMMA.
// LDS: sg   = raw filter, 128-short zero front pad (negative lags)
//      pg   = packed pairs pg[k] = sg[k] | (sg[k-1]<<16)  -> 1 ds_load_b32
//             per A-fragment VGPR
//      su   = u data [col 0..7][m 0..4095]
// Each wave owns 4 consecutive 16-row output tiles (4 accs) and shares one
// B fragment per 32-wide k-step across 4 WMMAs.
// ---------------------------------------------------------------------------
#define SG_SHORTS (PAD_ + MH_)              // 4224
#define PG_WORDS  (PAD_ + MH_)              // 4224 u32
#define SU_SHORTS (8 * MH_ + 64)
#define SG_BYTES  (SG_SHORTS * 2)           // 8448
#define PG_BYTES  (PG_WORDS * 4)            // 16896
#define SU_BYTES  (SU_SHORTS * 2)           // 65664
#define SMEM_BYTES (SG_BYTES + PG_BYTES + SU_BYTES)   // 91008

__global__ __launch_bounds__(256) void stu_conv_kernel(
    const unsigned short* __restrict__ g_even,  // [D_][MH_]
    const unsigned short* __restrict__ u_perm,  // [D_][8][MH_]
    float* __restrict__ out)                    // [B_, L_, D_] fp32
{
  extern __shared__ unsigned char smem[];
  unsigned short* sg = (unsigned short*)smem;
  unsigned int*   pg = (unsigned int*)(smem + SG_BYTES);
  unsigned short* su = (unsigned short*)(smem + SG_BYTES + PG_BYTES);

  const int d    = blockIdx.x;
  const int tid  = threadIdx.x;
  const int lane = tid & 31;
  const int w    = tid >> 5;

  // ---- stage g and u into LDS (vectorized b128) ----
  {
    const uint4* srcg = (const uint4*)(g_even + (size_t)d * MH_);
    uint4* dstg = (uint4*)(sg + PAD_);
#pragma unroll
    for (int j = 0; j < 2; ++j) dstg[tid + j * 256] = srcg[tid + j * 256];
    const uint4* srcu = (const uint4*)(u_perm + (size_t)d * (8 * MH_));
    uint4* dstu = (uint4*)su;
#pragma unroll
    for (int j = 0; j < 16; ++j) dstu[tid + j * 256] = srcu[tid + j * 256];
    if (tid < 64) ((unsigned int*)sg)[tid] = 0u;        // negative-lag zeros
    if (tid < 32) ((unsigned int*)(su + 8 * MH_))[tid] = 0u;  // tail pad
  }
  // warm next channel's u in L2 while we compute this one
  if (d + 1 < D_) {
    const char* nxt = (const char*)(u_perm + (size_t)(d + 1) * (8 * MH_));
    for (int j = tid; j < 512; j += 256)
      __builtin_prefetch(nxt + j * 128, 0, 0);
  }
  __syncthreads();

  // ---- build packed Toeplitz pairs: pg[k] = (g[k-1]<<16) | g[k] ----
  for (int k = tid; k < PG_WORDS; k += 256) {
    unsigned int lo = sg[k];
    unsigned int hi = (k > 0) ? (unsigned int)sg[k - 1] : 0u;
    pg[k] = lo | (hi << 16);
  }
  __syncthreads();

  const int kbA = (lane >= 16) ? 8 : 0;
  const int kbB = (lane >= 16) ? 16 : 0;
  const int p   = lane & 15;                 // A row within tile / B column

  for (int grp = w; grp < NTILE_ / 4; grp += 8) {
    const int i0 = grp << 2;                 // 4 consecutive i-tiles
    const int tb0 = i0 << 4;
    v8f acc[4];
#pragma unroll
    for (int s = 0; s < 4; ++s)
#pragma unroll
      for (int v = 0; v < 8; ++v) acc[s][v] = 0.f;

    const int njj = ((tb0 + 63) >> 5) + 1;   // k-range for largest tile
    for (int jj = 0; jj < njj; ++jj) {
      const int mbase = jj << 5;
      // shared u B fragment: col = p (batch*2+parity), zero cols 8..15
      FragAB bfr;
      if (p < 8) {
        const unsigned short* urow = su + (p << 12) + mbase + kbB;
#pragma unroll
        for (int v = 0; v < 8; ++v)
          bfr.u[v] = *(const unsigned int*)(urow + 2 * v);
      } else {
#pragma unroll
        for (int v = 0; v < 8; ++v) bfr.u[v] = 0u;
      }
      // 4 Toeplitz A fragments, one ds_load_b32 per VGPR from packed pairs
      const unsigned int* pgl = pg + PAD_ + tb0 + p - mbase - kbB;
#pragma unroll
      for (int s = 0; s < 4; ++s) {
        FragAB afr;
#pragma unroll
        for (int v = 0; v < 8; ++v) {
          int koff = 2 * v;
          afr.u[v] = pgl[(s << 4) - koff];
        }
        acc[s] = __builtin_amdgcn_wmma_f32_16x16x32_bf16(
            false, afr.v, false, bfr.v, (short)0, acc[s], false, false);
      }
    }

    // store: lane N=p -> (b,r); VGPR v -> half-rate row; t = 2m+r
    if (p < 8) {
      const int b = p >> 1, r = p & 1;
      float* obase = out + (size_t)b * L_ * D_ + d;
#pragma unroll
      for (int s = 0; s < 4; ++s) {
        const int rowbase = tb0 + (s << 4) + ((lane >> 4) << 3);
#pragma unroll
        for (int v = 0; v < 8; ++v) {
          int t = ((rowbase + v) << 1) | r;
          obase[(size_t)t * D_] = acc[s][v];
        }
      }
    }
  }
}

// ---------------------------------------------------------------------------
extern "C" void kernel_launch(void* const* d_in, const int* in_sizes, int n_in,
                              void* d_out, int out_size, void* d_ws, size_t ws_size,
                              hipStream_t stream) {
  const float* x   = (const float*)d_in[0];   // [B,L,D]
  const float* phi = (const float*)d_in[1];   // [L,K]
  const float* Mi  = (const float*)d_in[2];   // [D,D]
  const float* Mf  = (const float*)d_in[3];   // [K,D]
  float* out = (float*)d_out;

  unsigned short* g_even = (unsigned short*)d_ws;                       // 8 MB
  unsigned short* u_perm = (unsigned short*)((char*)d_ws + (8u << 20)); // 64 MB

  // filter projection + parity fold
  stu_filter_kernel<<<(D_ * MH_) / 256, 256, 0, stream>>>(phi, Mf, g_even);

  // input projection GEMM (WMMA) with fused parity permute
  dim3 gB(D_ / 64, (B_ * L_) / 128);
  stu_xproj_kernel<<<gB, 256, 0, stream>>>(x, Mi, u_perm);

  // per-channel Toeplitz conv (WMMA) — needs >64KB dynamic LDS
  hipFuncSetAttribute((const void*)stu_conv_kernel,
                      hipFuncAttributeMaxDynamicSharedMemorySize, SMEM_BYTES);
  stu_conv_kernel<<<D_, 256, SMEM_BYTES, stream>>>(g_even, u_perm, out);
}
